// MultiHeadAttention_52089363366285
// MI455X (gfx1250) — matrix operand order repaired and checked by reading, not compiled, not run on hardware
//
#include <hip/hip_runtime.h>

// ---------------- problem constants ----------------
#define NN 2048
#define DD 2048
#define HEADS 16
#define HDIM 128
#define ROUNDS 2

typedef __attribute__((ext_vector_type(16))) __bf16 v16bf;
typedef __attribute__((ext_vector_type(8)))  __bf16 v8bf;
typedef __attribute__((ext_vector_type(8)))  float  v8f;
typedef __attribute__((ext_vector_type(4))) unsigned int tdm_g0_t;
typedef __attribute__((ext_vector_type(8))) int          tdm_g1_t;
typedef __attribute__((ext_vector_type(4))) int          tdm_g2_t;

static __device__ inline v16bf cat8(v8bf lo, v8bf hi) {
  return __builtin_shufflevector(lo, hi, 0,1,2,3,4,5,6,7,8,9,10,11,12,13,14,15);
}
static __device__ inline v8f wmma_bf16(v16bf a, v16bf b, v8f c) {
  return __builtin_amdgcn_wmma_f32_16x16x32_bf16(false, a, false, b, (short)0, c, false, false);
}

#if defined(__gfx1250__) && __has_builtin(__builtin_amdgcn_tensor_load_to_lds)
#define USE_TDM 1
#else
#define USE_TDM 0
#endif

static __device__ inline void wait_tensorcnt0() {
#if defined(__gfx1250__)
#if __has_builtin(__builtin_amdgcn_s_wait_tensorcnt)
  __builtin_amdgcn_s_wait_tensorcnt(0);
#else
  asm volatile("s_wait_tensorcnt 0x0" ::: "memory");
#endif
#endif
}

#if USE_TDM
// Issue a TDM 2D-tile load: tile (elemsPerRow x rows) of bf16 from a
// (tensorCols x tensorRows) row-major tensor into LDS at ldsOff.
// D# pad feature: every 16 DWORDs (64B = 32 halves) insert 4 DWORDs (16B)
// -> LDS row stride of 40 halves, matching the compute-side layout.
static __device__ inline void tdm_load_tile_bf16(const __bf16* gtile,
                                                 unsigned ldsOff,
                                                 int rows, int elemsPerRow,
                                                 int tensorRows, int tensorCols) {
  unsigned long long ga = (unsigned long long)(uintptr_t)gtile;
  tdm_g0_t g0;
  g0[0] = 1u;                                   // count=1 (valid), user mode
  g0[1] = ldsOff;                               // lds_addr (bytes)
  g0[2] = (unsigned)(ga & 0xFFFFFFFFu);         // global_addr[31:0]
  g0[3] = (unsigned)((ga >> 32) & 0x1FFFFFFu)   // global_addr[56:32]
          | (2u << 30);                         // type = 2 (image)
  tdm_g1_t g1;
  g1[0] = (int)((1u << 16)                      // data_size = 1 -> 2 bytes
                | (1u << 20)                    // pad_enable
                | (3u << 22)                    // pad_interval: 16 DWORDs
                | (3u << 25));                  // pad_amount:   4 DWORDs
  g1[1] = (int)(((unsigned)tensorCols & 0xFFFFu) << 16);            // dim0 lo
  g1[2] = (int)((((unsigned)tensorCols >> 16) & 0xFFFFu)
                | (((unsigned)tensorRows & 0xFFFFu) << 16));        // dim0 hi | dim1 lo
  g1[3] = (int)((((unsigned)tensorRows >> 16) & 0xFFFFu)
                | (((unsigned)elemsPerRow & 0xFFFFu) << 16));       // dim1 hi | tile_dim0
  g1[4] = rows;                                 // tile_dim1 (tile_dim2 = 0)
  g1[5] = tensorCols;                           // tensor_dim0_stride lo
  g1[6] = 0;
  g1[7] = 0;
  tdm_g2_t z4 = {};
  tdm_g1_t z8 = {};
  // 6-arg form (clang-23 / therock-10.0 headers)
  __builtin_amdgcn_tensor_load_to_lds(g0, g1, z4, z4, z8, 0);
}
#endif

// ---------------- fp32 -> bf16 cast ----------------
__global__ void cast_f32_bf16_kernel(const float* __restrict__ src,
                                     __bf16* __restrict__ dst, int n) {
  int i = blockIdx.x * blockDim.x + threadIdx.x;
  int stride = gridDim.x * blockDim.x;
  for (; i < n; i += stride) dst[i] = (__bf16)src[i];
}

// ---------------- bf16 transpose (for V) ----------------
__global__ void transpose_bf16_kernel(const __bf16* __restrict__ src,
                                      __bf16* __restrict__ dst,
                                      int rows, int cols) {
  __shared__ __bf16 tile[32][33];
  int bx = blockIdx.x * 32, by = blockIdx.y * 32;
  int x = threadIdx.x, y = threadIdx.y;          // block (32, 8)
  for (int i = 0; i < 32; i += 8)
    tile[y + i][x] = src[(size_t)(by + y + i) * cols + bx + x];
  __syncthreads();
  for (int i = 0; i < 32; i += 8)
    dst[(size_t)(bx + y + i) * rows + by + x] = tile[x][y + i];
}

// ---------------- WMMA GEMM: out = act( A @ W^T + bias ) [+ resid] ----------
#define BM 128
#define BN 128
#define TK 32
#define LSTR 40   // LDS row stride in halves (32 data + 8 pad)

template <bool RELU, bool HAS_RES, bool OUT_F, bool OUT_B>
__global__ __launch_bounds__(256)
void gemm_bf16_kernel(const __bf16* __restrict__ A, const __bf16* __restrict__ W,
                      const float* __restrict__ bias,
                      const float* __restrict__ resid,
                      float* __restrict__ outF, __bf16* __restrict__ outB,
                      int M, int Nout, int K) {
  __shared__ __align__(16) __bf16 As[2][BM * LSTR];
  __shared__ __align__(16) __bf16 Bs[2][BN * LSTR];

  const int tid  = threadIdx.x;
  const int lane = tid & 31;
  const int wid  = tid >> 5;
  const int mBlock = blockIdx.y * BM;
  const int nBlock = blockIdx.x * BN;
  const int wm = (wid >> 1) * 32;
  const int wn = (wid & 1) * 64;

  const int mrow = lane & 15;
  const int hi8  = (lane >> 4) * 8;   // A-layout per-lane k base
  const int bkb  = (lane >> 4) * 16;  // unused; B fragments use hi8 like A

  v8f acc[2][4] = {};

  auto compute = [&](int buf) {
    v16bf afrag[2], bfrag[4];
    for (int tm = 0; tm < 2; ++tm) {
      const __bf16* p = &As[buf][(wm + tm * 16 + mrow) * LSTR];
      afrag[tm] = cat8(*(const v8bf*)(p + hi8), *(const v8bf*)(p + 16 + hi8));
    }
    for (int tn = 0; tn < 4; ++tn) {
      const __bf16* p = &Bs[buf][(wn + tn * 16 + mrow) * LSTR + hi8];
      bfrag[tn] = cat8(*(const v8bf*)(p), *(const v8bf*)(p + 16));
    }
    for (int tm = 0; tm < 2; ++tm)
      for (int tn = 0; tn < 4; ++tn)
        acc[tm][tn] = wmma_bf16(afrag[tm], bfrag[tn], acc[tm][tn]);
  };

  const int iters = K / TK;

#if USE_TDM
  // Stage tiles with the Tensor Data Mover: wave 0 issues two 2D-tile DMA
  // descriptors per K-step; TENSORcnt + workgroup barrier publish the LDS.
  auto issue = [&](int it, int buf) {
    const int k0 = it * TK;
    tdm_load_tile_bf16(A + (size_t)mBlock * K + k0,
                       (unsigned)(uintptr_t)&As[buf][0], BM, TK, M, K);
    tdm_load_tile_bf16(W + (size_t)nBlock * K + k0,
                       (unsigned)(uintptr_t)&Bs[buf][0], BN, TK, Nout, K);
  };
  if (wid == 0) {
    issue(0, 0);
    wait_tensorcnt0();
  }
  __syncthreads();
  for (int it = 0; it < iters; ++it) {
    const int buf = it & 1;
    if (it + 1 < iters && wid == 0) issue(it + 1, buf ^ 1);
    compute(buf);
    if (it + 1 < iters) {
      if (wid == 0) wait_tensorcnt0();
      __syncthreads();
    }
  }
#else
  const int c0 = tid, c1 = tid + 256;
  const int r0 = c0 >> 2, s0 = c0 & 3;
  const int r1 = c1 >> 2, s1 = c1 & 3;
  uint4 ga0, ga1, gb0, gb1;
  auto loadGlobal = [&](int k0) {
    ga0 = *(const uint4*)(A + (size_t)(mBlock + r0) * K + k0 + s0 * 8);
    ga1 = *(const uint4*)(A + (size_t)(mBlock + r1) * K + k0 + s1 * 8);
    gb0 = *(const uint4*)(W + (size_t)(nBlock + r0) * K + k0 + s0 * 8);
    gb1 = *(const uint4*)(W + (size_t)(nBlock + r1) * K + k0 + s1 * 8);
  };
  auto storeLds = [&](int buf) {
    *(uint4*)&As[buf][r0 * LSTR + s0 * 8] = ga0;
    *(uint4*)&As[buf][r1 * LSTR + s1 * 8] = ga1;
    *(uint4*)&Bs[buf][r0 * LSTR + s0 * 8] = gb0;
    *(uint4*)&Bs[buf][r1 * LSTR + s1 * 8] = gb1;
  };
  loadGlobal(0);
  storeLds(0);
  for (int it = 0; it < iters; ++it) {
    __syncthreads();
    const int buf = it & 1;
    if (it + 1 < iters) loadGlobal((it + 1) * TK);
    compute(buf);
    if (it + 1 < iters) storeLds(buf ^ 1);
  }
#endif

  // epilogue: bias + (relu) + (residual); branch-free per instantiation
  for (int tn = 0; tn < 4; ++tn) {
    const int ng = nBlock + wn + tn * 16 + mrow;
    const float bv = bias[ng];
    for (int tm = 0; tm < 2; ++tm) {
      const int mg0 = mBlock + wm + tm * 16 + hi8;
      for (int r = 0; r < 8; ++r) {
        float v = acc[tm][tn][r] + bv;
        if (RELU) v = fmaxf(v, 0.f);
        const size_t idx = (size_t)(mg0 + r) * Nout + ng;
        if (HAS_RES) v += resid[idx];
        if (OUT_F) outF[idx] = v;
        if (OUT_B) outB[idx] = (__bf16)v;
      }
    }
  }
}

// ---------------- flash attention (per head, online softmax) ---------------
__global__ __launch_bounds__(128)
void flash_attn_kernel(const __bf16* __restrict__ Q,
                       const __bf16* __restrict__ Km,
                       const __bf16* __restrict__ VT,
                       __bf16* __restrict__ O, float scale) {
  const int lane = threadIdx.x & 31;
  const int wid  = threadIdx.x >> 5;
  const int h    = blockIdx.y;
  const int hoff = h * HDIM;
  const int qbase = blockIdx.x * 64 + wid * 16;
  const int mrow = lane & 15;
  const int hi8  = (lane >> 4) * 8;
  const int bkb  = (lane >> 4) * 16;

  v16bf qf[4];
  {
    const __bf16* qp = Q + (size_t)(qbase + mrow) * DD + hoff;
    for (int c = 0; c < 4; ++c)
      qf[c] = cat8(*(const v8bf*)(qp + c * 32 + hi8),
                   *(const v8bf*)(qp + c * 32 + 16 + hi8));
  }

  float m_run = -1e30f, l_run = 0.f;
  v8f oacc[8] = {};

  for (int kv0 = 0; kv0 < NN; kv0 += 32) {
    v8f s[2] = {};
    for (int t = 0; t < 2; ++t) {
      const __bf16* kp = Km + (size_t)(kv0 + t * 16 + mrow) * DD + hoff;
      for (int c = 0; c < 4; ++c) {
        v16bf kf = cat8(*(const v8bf*)(kp + c * 32 + hi8),
                        *(const v8bf*)(kp + c * 32 + 16 + hi8));
        s[t] = wmma_bf16(kf, qf[c], s[t]);
      }
    }
    float bm = -1e30f;
    for (int t = 0; t < 2; ++t)
      for (int r = 0; r < 8; ++r) {
        s[t][r] *= scale;
        bm = fmaxf(bm, s[t][r]);
      }
    bm = fmaxf(bm, __shfl_xor(bm, 16, 32));
    const float m_new = fmaxf(m_run, bm);
    const float f = __expf(m_run - m_new);
    float rs = 0.f;
    v16bf pf;
    for (int r = 0; r < 8; ++r) {
      const float p0 = __expf(s[0][r] - m_new);
      const float p1 = __expf(s[1][r] - m_new);
      rs += p0 + p1;
      pf[r]     = (__bf16)p0;
      pf[r + 8] = (__bf16)p1;
    }
    rs += __shfl_xor(rs, 16, 32);
    l_run = l_run * f + rs;
    m_run = m_new;
    float fr[8];
    for (int r = 0; r < 8; ++r) fr[r] = __shfl(f, r + hi8, 32);
    for (int t2 = 0; t2 < 8; ++t2)
      for (int r = 0; r < 8; ++r) oacc[t2][r] *= fr[r];
    for (int t2 = 0; t2 < 8; ++t2) {
      const __bf16* vp = VT + (size_t)(hoff + t2 * 16 + mrow) * NN + kv0 + hi8;
      oacc[t2] = wmma_bf16(pf, cat8(*(const v8bf*)(vp), *(const v8bf*)(vp + 16)),
                           oacc[t2]);
    }
  }

  const float inv = 1.f / l_run;
  float lr[8];
  for (int r = 0; r < 8; ++r) lr[r] = __shfl(inv, r + hi8, 32);
  for (int t2 = 0; t2 < 8; ++t2)
    for (int r = 0; r < 8; ++r)
      O[(size_t)(qbase + r + hi8) * DD + hoff + t2 * 16 + mrow] =
          (__bf16)(oacc[t2][r] * lr[r]);
}

// ---------------- host orchestration ----------------
extern "C" void kernel_launch(void* const* d_in, const int* in_sizes, int n_in,
                              void* d_out, int out_size, void* d_ws, size_t ws_size,
                              hipStream_t stream) {
  (void)in_sizes; (void)n_in; (void)out_size; (void)ws_size;
  const float* ent = (const float*)d_in[0];
  const float* Wq = (const float*)d_in[1];  const float* bq = (const float*)d_in[2];
  const float* Wk = (const float*)d_in[3];  const float* bk = (const float*)d_in[4];
  const float* Wv = (const float*)d_in[5];  const float* bv = (const float*)d_in[6];
  const float* W0 = (const float*)d_in[7];  const float* b0 = (const float*)d_in[8];
  const float* W1 = (const float*)d_in[9];  const float* b1 = (const float*)d_in[10];

  const size_t NE = (size_t)NN * DD;
  char* ws = (char*)d_ws;
  size_t off = 0;
  auto alloc = [&](size_t bytes) -> void* {
    void* p = ws + off;
    off += (bytes + 255) & ~(size_t)255;
    return p;
  };
  __bf16* entB = (__bf16*)alloc(NE * 2);
  __bf16* WqB  = (__bf16*)alloc(NE * 2);
  __bf16* WkB  = (__bf16*)alloc(NE * 2);
  __bf16* WvB  = (__bf16*)alloc(NE * 2);
  __bf16* W0B  = (__bf16*)alloc(NE * 2);
  __bf16* W1B  = (__bf16*)alloc(NE * 2);
  __bf16* qB   = (__bf16*)alloc(NE * 2);
  __bf16* kB   = (__bf16*)alloc(NE * 2);
  __bf16* vB   = (__bf16*)alloc(NE * 2);
  __bf16* vT   = (__bf16*)alloc(NE * 2);
  __bf16* oB   = (__bf16*)alloc(NE * 2);
  __bf16* h0B  = (__bf16*)alloc(NE * 2);
  float*  entF = (float*)alloc(NE * 4);

  const int CB = 256, CG = 2048;
  cast_f32_bf16_kernel<<<CG, CB, 0, stream>>>(ent, entB, (int)NE);
  cast_f32_bf16_kernel<<<CG, CB, 0, stream>>>(Wq, WqB, (int)NE);
  cast_f32_bf16_kernel<<<CG, CB, 0, stream>>>(Wk, WkB, (int)NE);
  cast_f32_bf16_kernel<<<CG, CB, 0, stream>>>(Wv, WvB, (int)NE);
  cast_f32_bf16_kernel<<<CG, CB, 0, stream>>>(W0, W0B, (int)NE);
  cast_f32_bf16_kernel<<<CG, CB, 0, stream>>>(W1, W1B, (int)NE);
  (void)hipMemcpyAsync(entF, ent, NE * sizeof(float), hipMemcpyDeviceToDevice,
                       stream);

  const dim3 gGemm(NN / BN, NN / BM), bGemm(256);
  const dim3 gTr(DD / 32, NN / 32), bTr(32, 8);
  const dim3 gAtt(NN / 64, HEADS), bAtt(128);
  const float scale = 0.08838834764831845f;  // 1/sqrt(128)

  for (int round = 0; round < ROUNDS; ++round) {
    gemm_bf16_kernel<false, false, false, true><<<gGemm, bGemm, 0, stream>>>(
        entB, WqB, bq, nullptr, nullptr, qB, NN, DD, DD);
    gemm_bf16_kernel<false, false, false, true><<<gGemm, bGemm, 0, stream>>>(
        entB, WkB, bk, nullptr, nullptr, kB, NN, DD, DD);
    gemm_bf16_kernel<false, false, false, true><<<gGemm, bGemm, 0, stream>>>(
        entB, WvB, bv, nullptr, nullptr, vB, NN, DD, DD);
    transpose_bf16_kernel<<<gTr, bTr, 0, stream>>>(vB, vT, NN, DD);
    flash_attn_kernel<<<gAtt, bAtt, 0, stream>>>(qB, kB, vT, oB, scale);
    gemm_bf16_kernel<true, false, false, true><<<gGemm, bGemm, 0, stream>>>(
        oB, W0B, b0, nullptr, nullptr, h0B, NN, DD, DD);
    if (round == ROUNDS - 1) {
      gemm_bf16_kernel<true, true, true, false><<<gGemm, bGemm, 0, stream>>>(
          h0B, W1B, b1, entF, (float*)d_out, nullptr, NN, DD, DD);
    } else {
      gemm_bf16_kernel<true, true, true, true><<<gGemm, bGemm, 0, stream>>>(
          h0B, W1B, b1, entF, entF, entB, NN, DD, DD);
    }
  }
}
